// CriticCVaR_2259152798443
// MI455X (gfx1250) — compile-verified, matrix operand order checked by reading, not run on hardware
//
#include <hip/hip_runtime.h>

// MI455X / gfx1250, wave32. Fused: states@W1 + b1 -> SiLU -> @W2 + b2.
// bf16 split-precision (hi/lo) WMMA, fp32 accumulate. Memory-bound design:
// 273 MB HBM floor ~11.7us; 3x bf16 WMMA work stays under that ceiling.

typedef __attribute__((ext_vector_type(16))) __bf16 v16bf;
typedef __attribute__((ext_vector_type(8)))  float  v8f;
typedef __attribute__((ext_vector_type(4)))  float  f4;

#define T_ 32
#define B_ 8192
#define S_ 256
#define H_ 128

static __device__ __forceinline__ v8f wmma_bf16(v16bf a, v16bf b, v8f c) {
    return __builtin_amdgcn_wmma_f32_16x16x32_bf16(
        /*neg_a=*/false, a, /*neg_b=*/false, b,
        /*c_mod=*/(short)0, c, /*reuse_a=*/false, /*reuse_b=*/false);
}

__global__ __launch_bounds__(256) void critic_fused_kernel(
    const float* __restrict__ states,  // [T,B,S]
    const float* __restrict__ W1,      // [T,S,H]
    const float* __restrict__ b1,      // [T,H]
    const float* __restrict__ W2,      // [T,H] (H x 1)
    const float* __restrict__ b2,      // [T]
    float* __restrict__ out)           // [T,B]
{
    // Swizzled B-operand storage: idx = ((nt*8 + kb)*32 + lane)*16 + e
    // nt: N-tile (H/16), kb: K-chunk (S/32), lane: consuming lane, e: bf16 elem.
    __shared__ __bf16 sBhi[8 * 8 * 32 * 16];   // 64 KB
    __shared__ __bf16 sBlo[8 * 8 * 32 * 16];   // 64 KB

    const int tid  = threadIdx.x;
    const int lane = tid & 31;
    const int wave = tid >> 5;
    const int t    = blockIdx.x >> 6;          // 64 blocks per t
    const int mblk = blockIdx.x & 63;
    const int m0   = mblk * 128 + wave * 16;   // this wave's 16 rows of B

    const float* W1t = W1 + (size_t)t * (S_ * H_);

    // ---- Stage W1[t] into LDS, split fp32 -> bf16 hi/lo, pre-swizzled ----
    // Coalesced float4 reads over [K=256][col=128]; scatter 2B stores to LDS.
    for (int s4 = tid; s4 < (S_ * H_ / 4); s4 += 256) {
        const int K  = s4 >> 5;            // 0..255
        const int c0 = (s4 & 31) * 4;      // 0..124 (never crosses an N-tile)
        f4 v = *(const f4*)(W1t + K * H_ + c0);
        const int kb  = K >> 5;
        const int kr  = K & 31;
        const int g   = kr >> 4;
        const int hh  = (kr >> 3) & 1;
        const int e   = (kr & 7) | (g << 3);
        const int nt  = c0 >> 4;
        const int base = (((nt * 8 + kb) * 32 + hh * 16) * 16) + e;
        #pragma unroll
        for (int j = 0; j < 4; ++j) {
            const float  x  = v[j];
            const __bf16 xh = (__bf16)x;
            const __bf16 xl = (__bf16)(x - (float)xh);
            const int    n  = (c0 + j) & 15;
            sBhi[base + n * 16] = xh;
            sBlo[base + n * 16] = xl;
        }
    }
    __syncthreads();

    // ---- Preload this wave's A tile (16 rows x 256 K) as split bf16 regs ----
    // 16-bit A layout: lane -> (M = lane&15, half = lane>>4);
    // element e -> K = 16*(e>>3) + 8*half + (e&7).
    const int m  = lane & 15;
    const int hh = lane >> 4;
    const float* arow = states + ((size_t)t * B_ + m0 + m) * S_;

    v16bf ahi[8], alo[8];
    #pragma unroll
    for (int kb = 0; kb < 8; ++kb) {
        const int c = kb * 32 + hh * 8;
        f4 r0 = *(const f4*)(arow + c);
        f4 r1 = *(const f4*)(arow + c + 4);
        f4 r2 = *(const f4*)(arow + c + 16);
        f4 r3 = *(const f4*)(arow + c + 20);
        #pragma unroll
        for (int e = 0; e < 16; ++e) {
            const float x = (e < 4)  ? r0[e]
                          : (e < 8)  ? r1[e - 4]
                          : (e < 12) ? r2[e - 8]
                          :            r3[e - 12];
            const __bf16 xh = (__bf16)x;
            ahi[kb][e] = xh;
            alo[kb][e] = (__bf16)(x - (float)xh);
        }
    }

    // ---- Main loop: 8 N-tiles x 8 K-chunks x 3 split-product WMMAs ----
    const float* b1t = b1 + t * H_;
    const float* w2t = W2 + t * H_;
    const float  b2v = b2[t];

    float rowsum[8] = {0.f, 0.f, 0.f, 0.f, 0.f, 0.f, 0.f, 0.f};
    const v16bf* vBhi = (const v16bf*)sBhi;
    const v16bf* vBlo = (const v16bf*)sBlo;

    for (int nt = 0; nt < 8; ++nt) {
        v8f c = {};
        #pragma unroll
        for (int kb = 0; kb < 8; ++kb) {
            const int bi = (nt * 8 + kb) * 32 + lane;
            v16bf bh = vBhi[bi];
            v16bf bl = vBlo[bi];
            c = wmma_bf16(ahi[kb], bh, c);  // hi*hi
            c = wmma_bf16(ahi[kb], bl, c);  // hi*lo
            c = wmma_bf16(alo[kb], bh, c);  // lo*hi
        }
        // C/D layout: VGPR r -> M = 8*hh + r, N = lane&15 (column broadcast).
        const float b1v = b1t[nt * 16 + (lane & 15)];
        const float w2v = w2t[nt * 16 + (lane & 15)];
        #pragma unroll
        for (int r = 0; r < 8; ++r) {
            const float x = c[r] + b1v;
            const float s = x * __builtin_amdgcn_rcpf(1.0f + __expf(-x)); // SiLU
            rowsum[r] += s * w2v;
        }
    }

    // ---- Reduce across the 16 lanes holding one N-strip each ----
    #pragma unroll
    for (int r = 0; r < 8; ++r) {
        float v = rowsum[r];
        v += __shfl_xor(v, 1, 32);
        v += __shfl_xor(v, 2, 32);
        v += __shfl_xor(v, 4, 32);
        v += __shfl_xor(v, 8, 32);
        if ((lane & 15) == 0) {
            out[(size_t)t * B_ + m0 + hh * 8 + r] = v + b2v;
        }
    }
}

extern "C" void kernel_launch(void* const* d_in, const int* in_sizes, int n_in,
                              void* d_out, int out_size, void* d_ws, size_t ws_size,
                              hipStream_t stream) {
    (void)in_sizes; (void)n_in; (void)out_size; (void)d_ws; (void)ws_size;
    const float* states = (const float*)d_in[0];
    const float* W1     = (const float*)d_in[1];
    const float* b1     = (const float*)d_in[2];
    const float* W2     = (const float*)d_in[3];
    const float* b2     = (const float*)d_in[4];
    float* out          = (float*)d_out;

    dim3 grid(T_ * (B_ / 128));   // 2048 blocks: (t, 128-row block)
    dim3 block(256);              // 8 wave32 waves
    critic_fused_kernel<<<grid, block, 0, stream>>>(states, W1, b1, W2, b2, out);
}
